// GPT2_MLA_29506425323771
// MI455X (gfx1250) — matrix operand order repaired and checked by reading, not compiled, not run on hardware
//
#include <hip/hip_runtime.h>
#include <hip/hip_bf16.h>
#include <math.h>

// Model dims
#define LYR    12
#define DM     768
#define VOCAB  50257
#define NH     12
#define HD_    64
#define DLAT   256
#define NB     2
#define NT     1024
#define BT     (NB * NT)

typedef float  v8f   __attribute__((ext_vector_type(8)));
typedef __bf16 v16bf __attribute__((ext_vector_type(16)));
typedef unsigned int u32x4 __attribute__((ext_vector_type(4)));
typedef int          i32x8 __attribute__((ext_vector_type(8)));
typedef int          i32x4 __attribute__((ext_vector_type(4)));

__device__ __forceinline__ unsigned short f2bf_u(float f) {
  unsigned u = __builtin_bit_cast(unsigned, f);
  u += 0x7FFFu + ((u >> 16) & 1u);   // round-to-nearest-even
  return (unsigned short)(u >> 16);
}
__device__ __forceinline__ __bf16 us2bf(unsigned short s) {
  return __builtin_bit_cast(__bf16, s);
}

__device__ __forceinline__ v8f wmma_bf16(v16bf a, v16bf b, v8f c) {
  return __builtin_amdgcn_wmma_f32_16x16x32_bf16(false, a, false, b, (short)0, c,
                                                 false, false);
}

// A fragment 16x32 (MxK), source row-major [16][stride] bf16-as-ushort.
// Lanes 0-15: M=lane, K {0..15}; lanes 16-31: M=lane-16, K {16..31}.
// Contiguous 8-half (16B) runs -> ds_load_b128 pairs.
__device__ __forceinline__ v16bf load_afrag(const unsigned short* base, int lane, int stride) {
  v16bf a;
  int m  = lane & 15;
  int kb = (lane < 16) ? 0 : 16;
#pragma unroll
  for (int v = 0; v < 8; ++v) {
#pragma unroll
    for (int e = 0; e < 2; ++e) {
      int kk = kb + 2 * v + e;
      a[v * 2 + e] = us2bf(base[m * stride + kk]);
    }
  }
  return a;
}

// B fragment 32x16 (KxN) from k-contiguous storage: element(k,n) at base[n*ld + k].
// 16 contiguous halves per lane -> 2x ds_load_b128.
__device__ __forceinline__ v16bf load_bfragT(const unsigned short* base, int lane, int ld) {
  v16bf b;
  int n  = lane & 15;
  int kb = (lane < 16) ? 0 : 16;
#pragma unroll
  for (int i = 0; i < 16; ++i) b[i] = us2bf(base[n * ld + kb + i]);
  return b;
}

// B fragment 32x16 (KxN) from n-contiguous storage: element(k,n) at base[k*ld + n].
__device__ __forceinline__ v16bf load_bfrag(const unsigned short* base, int lane, int ld) {
  v16bf b;
  int n  = lane & 15;
  int kb = (lane < 16) ? 0 : 16;
#pragma unroll
  for (int i = 0; i < 16; ++i) b[i] = us2bf(base[(kb + i) * ld + n]);
  return b;
}

// ---------------------------------------------------------------------------
// Embedding: x[bt,:] = wte[idx[bt],:] + wpe[t,:]   (fp32, feeds residual stream)
__global__ __launch_bounds__(256) void k_embed(const int* __restrict__ idx,
                                               const float* __restrict__ wte,
                                               const float* __restrict__ wpe,
                                               float* __restrict__ x) {
  int bt = blockIdx.x;
  int t  = bt % NT;
  int id = idx[bt];
#pragma unroll
  for (int j = 0; j < 3; ++j) {
    int d = threadIdx.x + j * 256;
    x[(size_t)bt * DM + d] = wte[(size_t)id * DM + d] + wpe[(size_t)t * DM + d];
  }
}

// ---------------------------------------------------------------------------
// LayerNorm fp32 -> bf16 (output only feeds WMMA GEMM A-operands).
__global__ __launch_bounds__(256) void k_ln(const float* __restrict__ in,
                                            unsigned short* __restrict__ out,
                                            const float* __restrict__ g,
                                            const float* __restrict__ be) {
  int r = blockIdx.x;
  const float* xp = in + (size_t)r * DM;
  int tid = threadIdx.x;
  float v0 = xp[tid], v1 = xp[tid + 256], v2 = xp[tid + 512];
  float s  = v0 + v1 + v2;
  float s2 = v0 * v0 + v1 * v1 + v2 * v2;
#pragma unroll
  for (int off = 16; off >= 1; off >>= 1) {
    s  += __shfl_xor(s,  off, 32);
    s2 += __shfl_xor(s2, off, 32);
  }
  __shared__ float red[16];
  int wv = tid >> 5, lane = tid & 31;
  if (lane == 0) { red[wv] = s; red[8 + wv] = s2; }
  __syncthreads();
  float ts = 0.f, ts2 = 0.f;
#pragma unroll
  for (int i = 0; i < 8; ++i) { ts += red[i]; ts2 += red[8 + i]; }
  float mean = ts * (1.0f / DM);
  float var  = ts2 * (1.0f / DM) - mean * mean;
  float rstd = rsqrtf(var + 1e-5f);
  unsigned short* op = out + (size_t)r * DM;
  op[tid]       = f2bf_u((v0 - mean) * rstd * g[tid]       + be[tid]);
  op[tid + 256] = f2bf_u((v1 - mean) * rstd * g[tid + 256] + be[tid + 256]);
  op[tid + 512] = f2bf_u((v2 - mean) * rstd * g[tid + 512] + be[tid + 512]);
}

// ---------------------------------------------------------------------------
// Fused RoPE on q and k-half of kv (fp32 in-place). 8 rows/block, 32 lanes/row.
__global__ __launch_bounds__(256) void k_rope(float* __restrict__ q,
                                              float* __restrict__ kv) {
  int rid = blockIdx.x * 8 + (threadIdx.x >> 5);
  int d   = threadIdx.x & 31;
  int b   = rid / (NT * NH);
  int rem = rid % (NT * NH);
  int t   = rem / NH;
  int h   = rem % NH;
  float inv = __powf(10000.0f, -(float)(2 * d) / (float)HD_);
  float ang = (float)t * inv;
  float c = __cosf(ang), sn = __sinf(ang);
  {
    float* qp = q + (size_t)(b * NT + t) * DM + h * HD_;
    float x1 = qp[d], x2 = qp[d + 32];
    qp[d]      = x1 * c - x2 * sn;
    qp[d + 32] = x2 * c + x1 * sn;
  }
  {
    float* kp = kv + (size_t)(b * NT + t) * (2 * DM) + h * HD_;
    float x1 = kp[d], x2 = kp[d + 32];
    kp[d]      = x1 * c - x2 * sn;
    kp[d + 32] = x2 * c + x1 * sn;
  }
}

// ---------------------------------------------------------------------------
// Tiled bf16 WMMA GEMM: C[M,N] = A_bf16[M,K] @ W_f32 + bias (+resid) (+gelu)
// Block tile 128x128, 8 waves (4 M-subtiles of 32 x 2 N-groups of 64), K step 32.
// A tile (128x32x2B) staged via Tensor Data Mover (wave 0 issues TENSOR_LOAD_TO_LDS,
// waits TENSORcnt); B tile staged fp32->bf16 transposed (k-contiguous) for b128 frags.
template <bool TRANSB, bool OUTBF>
__global__ __launch_bounds__(256) void gemm_kernel(const unsigned short* __restrict__ A,
                                                   const float* __restrict__ W,
                                                   const float* __restrict__ bias,
                                                   const float* __restrict__ resid,
                                                   void* __restrict__ Cv,
                                                   int M, int N, int K, int act_gelu) {
  __shared__ unsigned short a_s[128 * 32];   // [m][k]
  __shared__ unsigned short b_s[128 * 32];   // [n][k]  (transposed: k contiguous)
  int tid = threadIdx.x, wv = tid >> 5, lane = tid & 31;
  int wm = wv & 3, wn = wv >> 2;
  int m0 = blockIdx.y * 128;
  int n0 = blockIdx.x * 128;
  v8f acc[2][4] = {};

  // --- Tensor DMA descriptor (D#) for the A tile: 2D tensor [M rows x K cols] bf16,
  // tile 32 (dim0=K) x 128 (dim1=M). Groups per ISA 8.3/8.4; groups 2/3 zero (2D).
  unsigned lds_off = (unsigned)(unsigned long long)(const void*)a_s;
  u32x4 g0;
  i32x8 g1;
  i32x4 gz4;
  i32x8 gz8;
  gz4[0] = 0; gz4[1] = 0; gz4[2] = 0; gz4[3] = 0;
#pragma unroll
  for (int i = 0; i < 8; ++i) gz8[i] = 0;
  g1[0] = (1 << 16);                               // data_size = 1 (2 bytes)
  g1[1] = (int)((unsigned)(K & 0xFFFF) << 16);     // tensor_dim0[15:0]
  g1[2] = (int)(((unsigned)K >> 16) | ((unsigned)(M & 0xFFFF) << 16));
  g1[3] = (int)(((unsigned)M >> 16) | (32u << 16));      // tile_dim0 = 32
  g1[4] = 128;                                     // tile_dim1 = 128
  g1[5] = (int)(unsigned)K;                        // tensor_dim0_stride[31:0]
  g1[6] = 0;
  g1[7] = 0;
  g0[0] = 1;                                       // count = 1
  g0[1] = lds_off;                                 // lds_addr

  for (int k0 = 0; k0 < K; k0 += 32) {
    // A tile via TDM (wave 0 only; TDM ignores EXEC, one DMA per block)
    if (wv == 0) {
      unsigned long long gaddr =
          (unsigned long long)(const void*)(A + (size_t)m0 * K + k0);
      g0[2] = (unsigned)(gaddr & 0xFFFFFFFFu);
      g0[3] = (unsigned)((gaddr >> 32) & 0x1FFFFFFu) | (2u << 30);  // type=2
      __builtin_amdgcn_tensor_load_to_lds(g0, g1, gz4, gz4, gz8, 0);
      __builtin_amdgcn_s_wait_tensorcnt(0);
    }
    {  // stage B tile 32x128 fp32 -> bf16, stored transposed [n][k]
      int row = tid >> 3, cb = (tid & 7) * 16;
#pragma unroll
      for (int i = 0; i < 16; ++i) {
        int n = n0 + cb + i;
        float v = 0.0f;
        if (n < N)
          v = TRANSB ? W[(size_t)n * K + k0 + row] : W[(size_t)(k0 + row) * N + n];
        b_s[(cb + i) * 32 + row] = f2bf_u(v);
      }
      if (k0 + 32 < K) {  // prefetch next weight tile -> global_prefetch_b8
        int n = n0 + cb;
        if (n < N)
          __builtin_prefetch(TRANSB ? (const void*)&W[(size_t)n * K + k0 + 32 + row]
                                    : (const void*)&W[(size_t)(k0 + 32 + row) * N + n],
                             0, 1);
      }
    }
    __syncthreads();
    v16bf af0 = load_afrag(a_s + (wm * 32) * 32, lane, 32);
    v16bf af1 = load_afrag(a_s + (wm * 32 + 16) * 32, lane, 32);
#pragma unroll
    for (int nt = 0; nt < 4; ++nt) {
      v16bf bf = load_bfragT(b_s + (wn * 64 + nt * 16) * 32, lane, 32);
      acc[0][nt] = wmma_bf16(af0, bf, acc[0][nt]);
      acc[1][nt] = wmma_bf16(af1, bf, acc[1][nt]);
    }
    __syncthreads();
  }

  // Epilogue: bias -> gelu -> residual -> store (fp32 or bf16)
  float* Cf = (float*)Cv;
  unsigned short* Cb = (unsigned short*)Cv;
  int col  = lane & 15;
  int roff = (lane < 16) ? 0 : 8;
#pragma unroll
  for (int mi = 0; mi < 2; ++mi) {
#pragma unroll
    for (int nt = 0; nt < 4; ++nt) {
      int n = n0 + wn * 64 + nt * 16 + col;
      if (n < N) {
#pragma unroll
        for (int v = 0; v < 8; ++v) {
          int m = m0 + wm * 32 + mi * 16 + v + roff;
          float val = acc[mi][nt][v];
          if (bias) val += bias[n];
          if (act_gelu) val = 0.5f * val * (1.0f + erff(val * 0.70710678f));
          if (resid) val += resid[(size_t)m * N + n];
          if (OUTBF) Cb[(size_t)m * N + n] = f2bf_u(val);
          else       Cf[(size_t)m * N + n] = val;
        }
      }
    }
  }
}

// ---------------------------------------------------------------------------
// Flash attention (causal). Grid (T/128, B*H). 8 waves, 16 query rows per wave.
// S = Q K^T via 2 WMMAs (K-dim = HD 2x32); P V via 4 WMMAs (K-dim = 32 keys).
// Output bf16 (feeds out-proj GEMM A-operand).
__global__ __launch_bounds__(256) void k_flash(const float* __restrict__ q,
                                               const float* __restrict__ kvbuf,
                                               unsigned short* __restrict__ o) {
  __shared__ unsigned short kt[32 * 64];
  __shared__ unsigned short vt[32 * 64];
  __shared__ unsigned short pt[8][16 * 32];

  int tid = threadIdx.x, wv = tid >> 5, lane = tid & 31;
  int bh = blockIdx.y, b = bh / NH, h = bh % NH;
  int qblk = blockIdx.x * 128;
  int q0 = qblk + wv * 16;
  int mrowlane = lane & 15;
  int kb8 = (lane < 16) ? 0 : 8;
  int roff = (lane < 16) ? 0 : 8;

  // Q fragments (pre-scaled by 1/sqrt(HD)=0.125), hd 0..31 and 32..63
  v16bf qf0, qf1;
  {
    const float* qp = q + (size_t)(b * NT + q0 + mrowlane) * DM + h * HD_;
#pragma unroll
    for (int v = 0; v < 8; ++v) {
#pragma unroll
      for (int e = 0; e < 2; ++e) {
        int kk = ((lane < 16) ? 0 : 16) + 2 * v + e;
        qf0[v * 2 + e] = us2bf(f2bf_u(qp[kk] * 0.125f));
        qf1[v * 2 + e] = us2bf(f2bf_u(qp[32 + kk] * 0.125f));
      }
    }
  }

  float mrow[8], lrow[8];
  v8f oacc[4] = {};
#pragma unroll
  for (int v = 0; v < 8; ++v) { mrow[v] = -1e30f; lrow[v] = 0.0f; }

  for (int ks = 0; ks < qblk + 128; ks += 32) {
    {  // cooperative stage of K,V tiles [32 keys x 64 hd] as bf16
      int row = tid >> 3, cb = (tid & 7) * 8;
      const float* kp = kvbuf + (size_t)(b * NT + ks + row) * (2 * DM) + h * HD_ + cb;
      const float* vp = kp + DM;
#pragma unroll
      for (int i = 0; i < 8; ++i) kt[row * 64 + cb + i] = f2bf_u(kp[i]);
#pragma unroll
      for (int i = 0; i < 8; ++i) vt[row * 64 + cb + i] = f2bf_u(vp[i]);
    }
    __syncthreads();

    bool active = (ks <= q0 + 15);  // wave-uniform
    float alpha[8];
    if (active) {
      // S = Q K^T : two key subtiles of 16, contraction over HD in 2 steps
      v8f s0 = {}, s1 = {};
      s0 = wmma_bf16(qf0, load_bfragT(kt, lane, 64), s0);
      s0 = wmma_bf16(qf1, load_bfragT(kt + 32, lane, 64), s0);
      s1 = wmma_bf16(qf0, load_bfragT(kt + 16 * 64, lane, 64), s1);
      s1 = wmma_bf16(qf1, load_bfragT(kt + 16 * 64 + 32, lane, 64), s1);

      // online softmax, row-wise across 16-lane groups
#pragma unroll
      for (int v = 0; v < 8; ++v) {
        int rown = q0 + v + roff;
        int key0 = ks + mrowlane, key1 = key0 + 16;
        float a = (key0 <= rown) ? s0[v] : -1e30f;
        float c = (key1 <= rown) ? s1[v] : -1e30f;
        float mx = fmaxf(a, c);
#pragma unroll
        for (int off = 8; off >= 1; off >>= 1) mx = fmaxf(mx, __shfl_xor(mx, off, 16));
        float mn = fmaxf(mrow[v], mx);
        float al = __expf(mrow[v] - mn);
        float p0 = __expf(a - mn);
        float p1 = __expf(c - mn);
        float rs = p0 + p1;
#pragma unroll
        for (int off = 8; off >= 1; off >>= 1) rs += __shfl_xor(rs, off, 16);
        lrow[v] = lrow[v] * al + rs;
        mrow[v] = mn;
        alpha[v] = al;
        pt[wv][(v + roff) * 32 + mrowlane]      = f2bf_u(p0);
        pt[wv][(v + roff) * 32 + 16 + mrowlane] = f2bf_u(p1);
      }
#pragma unroll
      for (int nt = 0; nt < 4; ++nt)
#pragma unroll
        for (int v = 0; v < 8; ++v) oacc[nt][v] *= alpha[v];
    }
    __syncthreads();
    if (active) {
      v16bf pf = load_afrag(pt[wv], lane, 32);
#pragma unroll
      for (int nt = 0; nt < 4; ++nt) {
        v16bf vf = load_bfrag(vt + nt * 16, lane, 64);
        oacc[nt] = wmma_bf16(pf, vf, oacc[nt]);
      }
    }
    __syncthreads();
  }

  // O = O / l  -> bf16
#pragma unroll
  for (int nt = 0; nt < 4; ++nt) {
#pragma unroll
    for (int v = 0; v < 8; ++v) {
      int rown = q0 + v + roff;
      float l = lrow[v];
      float val = (l > 0.0f) ? oacc[nt][v] / l : 0.0f;
      o[(size_t)(b * NT + rown) * DM + h * HD_ + nt * 16 + mrowlane] = f2bf_u(val);
    }
  }
}

// ---------------------------------------------------------------------------
extern "C" void kernel_launch(void* const* d_in, const int* in_sizes, int n_in,
                              void* d_out, int out_size, void* d_ws, size_t ws_size,
                              hipStream_t stream) {
  (void)in_sizes; (void)n_in; (void)out_size; (void)ws_size;

  const int*   idx    = (const int*)  d_in[0];
  const float* wte    = (const float*)d_in[1];
  const float* wpe    = (const float*)d_in[2];
  const float* ln1_w  = (const float*)d_in[3];
  const float* ln1_b  = (const float*)d_in[4];
  const float* wq_w   = (const float*)d_in[5];
  const float* wq_b   = (const float*)d_in[6];
  const float* kva_w  = (const float*)d_in[7];
  const float* kva_b  = (const float*)d_in[8];
  const float* kvb_w  = (const float*)d_in[9];
  const float* kvb_b  = (const float*)d_in[10];
  const float* out_w  = (const float*)d_in[11];
  const float* out_b  = (const float*)d_in[12];
  const float* ln2_w  = (const float*)d_in[13];
  const float* ln2_b  = (const float*)d_in[14];
  const float* fc_w   = (const float*)d_in[15];
  const float* fc_b   = (const float*)d_in[16];
  const float* proj_w = (const float*)d_in[17];
  const float* proj_b = (const float*)d_in[18];
  const float* lnf_w  = (const float*)d_in[19];
  const float* lnf_b  = (const float*)d_in[20];
  float* logits = (float*)d_out;

  // Workspace carve-up
  char* p = (char*)d_ws;
  float* x = (float*)p;                 p += (size_t)BT * DM * 4;      // residual
  unsigned short* hbuf = (unsigned short*)p; p += (size_t)BT * DM * 2; // LN out (bf16)
  float* qb = (float*)p;                p += (size_t)BT * DM * 4;      // q (fp32, rope)
  unsigned short* kvlat = (unsigned short*)p; p += (size_t)BT * DLAT * 2; // latent (bf16)
  float* kvb_a = (float*)p;             p += (size_t)BT * 2 * DM * 4;  // kv (fp32, rope)
  unsigned short* attnb = (unsigned short*)p; p += (size_t)BT * DM * 2;   // attn (bf16)
  unsigned short* fcb = (unsigned short*)p;   p += (size_t)BT * 4 * DM * 2; // gelu (bf16)

  dim3 blk(256);
  dim3 gLN(BT);
  dim3 gG768((DM + 127) / 128, BT / 128);
  dim3 gG256((DLAT + 127) / 128, BT / 128);
  dim3 gG1536((2 * DM + 127) / 128, BT / 128);
  dim3 gG3072((4 * DM + 127) / 128, BT / 128);
  dim3 gFlash(NT / 128, NB * NH);
  dim3 gRope((NB * NT * NH) / 8);
  dim3 gLogit((VOCAB + 127) / 128, BT / 128);

  k_embed<<<gLN, blk, 0, stream>>>(idx, wte, wpe, x);

  for (int l = 0; l < LYR; ++l) {
    k_ln<<<gLN, blk, 0, stream>>>(x, hbuf, ln1_w + (size_t)l * DM, ln1_b + (size_t)l * DM);

    gemm_kernel<false, false><<<gG768, blk, 0, stream>>>(
        hbuf, wq_w + (size_t)l * DM * DM, wq_b + (size_t)l * DM,
        nullptr, qb, BT, DM, DM, 0);

    gemm_kernel<false, true><<<gG256, blk, 0, stream>>>(
        hbuf, kva_w + (size_t)l * DM * DLAT, kva_b + (size_t)l * DLAT,
        nullptr, kvlat, BT, DLAT, DM, 0);

    gemm_kernel<false, false><<<gG1536, blk, 0, stream>>>(
        kvlat, kvb_w + (size_t)l * DLAT * 2 * DM, kvb_b + (size_t)l * 2 * DM,
        nullptr, kvb_a, BT, 2 * DM, DLAT, 0);

    k_rope<<<gRope, blk, 0, stream>>>(qb, kvb_a);

    k_flash<<<gFlash, blk, 0, stream>>>(qb, kvb_a, attnb);

    gemm_kernel<false, false><<<gG768, blk, 0, stream>>>(
        attnb, out_w + (size_t)l * DM * DM, out_b + (size_t)l * DM,
        x, x, BT, DM, DM, 0);

    k_ln<<<gLN, blk, 0, stream>>>(x, hbuf, ln2_w + (size_t)l * DM, ln2_b + (size_t)l * DM);

    gemm_kernel<false, true><<<gG3072, blk, 0, stream>>>(
        hbuf, fc_w + (size_t)l * DM * 4 * DM, fc_b + (size_t)l * 4 * DM,
        nullptr, fcb, BT, 4 * DM, DM, 1);

    gemm_kernel<false, false><<<gG768, blk, 0, stream>>>(
        fcb, proj_w + (size_t)l * 4 * DM * DM, proj_b + (size_t)l * DM,
        x, x, BT, DM, 4 * DM, 0);
  }

  k_ln<<<gLN, blk, 0, stream>>>(x, hbuf, lnf_w, lnf_b);

  // logits = h @ wte^T  (tied head, TRANSB)
  gemm_kernel<true, false><<<gLogit, blk, 0, stream>>>(
      hbuf, wte, nullptr, nullptr, logits, BT, VOCAB, DM, 0);
}
